// VoiceCloningTTS_3539053052144
// MI455X (gfx1250) — matrix operand order repaired and checked by reading, not compiled, not run on hardware
//
#include <hip/hip_runtime.h>
#include <hip/hip_bf16.h>
#include <math.h>

typedef __attribute__((ext_vector_type(8)))  float   v8f;
typedef __attribute__((ext_vector_type(16))) __bf16  v16bf;
typedef __attribute__((ext_vector_type(8)))  __bf16  v8bf;
typedef __attribute__((ext_vector_type(4)))  __bf16  v4bf;

__device__ __forceinline__ float sigf(float x) { return 1.f / (1.f + __expf(-x)); }

// --- Fragment builders (contiguous k = 16*half + s order, wave32), bf16 storage -----
// A(16x32): lane lm = row m; two contiguous 8-elem runs -> 2x 16B loads.
__device__ __forceinline__ v16bf frag_a_bf(const __bf16* row, int kb, int half) {
  const __bf16* p = row + kb + half * 16;
  const v8bf lo = *(const v8bf*)(p);
  const v8bf hi = *(const v8bf*)(p + 8);
  return __builtin_shufflevector(lo, hi, 0, 1, 2, 3, 4, 5, 6, 7,
                                 8, 9, 10, 11, 12, 13, 14, 15);
}
// B(32x16): lane lm = column n; 16 contiguous elems of W[n][*] -> one 32B load.
__device__ __forceinline__ v16bf frag_b_bf(const __bf16* p) {
  return *(const v16bf*)(p);
}
#define WMMA_BF16(a, b, acc) \
  __builtin_amdgcn_wmma_f32_16x16x32_bf16(false, (a), false, (b), (short)0, (acc), false, false)

// ---------------------------------------------------------------------------
// f32 -> bf16 row repack with zero pad: dst[r][0..Kpad) = bf16(src[r][0..K))|0
// ---------------------------------------------------------------------------
__global__ void pad_rows_bf16_kernel(const float* __restrict__ src, __bf16* __restrict__ dst,
                                     int K, int Kpad, int total) {
  const int idx = blockIdx.x * blockDim.x + threadIdx.x;
  if (idx >= total) return;
  const int r = idx / Kpad, k = idx - r * Kpad;
  dst[idx] = (k < K) ? (__bf16)src[r * K + k] : (__bf16)0.f;
}

// ---------------------------------------------------------------------------
// Embedding lookup -> bf16: et[(l*32+b)*256 + k] = bf16(emb[tok[b][l]][k])
// ---------------------------------------------------------------------------
__global__ void embed_kernel(const int* __restrict__ tokens,
                             const float* __restrict__ emb, __bf16* __restrict__ et) {
  const int idx = blockIdx.x * blockDim.x + threadIdx.x;   // over 512*32*64
  if (idx >= 512 * 32 * 64) return;
  const int k4  = idx & 63;
  const int row = idx >> 6;          // l*32 + b
  const int b   = row & 31;
  const int l   = row >> 5;
  const int tok = tokens[b * 512 + l];
  const float4 v = ((const float4*)(emb + (size_t)tok * 256))[k4];
  v4bf o;
  o[0] = (__bf16)v.x; o[1] = (__bf16)v.y; o[2] = (__bf16)v.z; o[3] = (__bf16)v.w;
  ((v4bf*)et)[idx] = o;
}

// ---------------------------------------------------------------------------
// Conv1d (pad=1,k=3) implicit-GEMM WMMA, bf16 weights (pre-padded to KPAD).
// K layout k = cin*3+dk (natural weight order). Input halo tile staged in LDS
// as bf16; B fragment gathers from LDS with compile-time folded guards.
// ---------------------------------------------------------------------------
template <int CIN>
__global__ __launch_bounds__(256)
void conv_kernel(const float* __restrict__ in_spk, const float* __restrict__ in_emo,
                 const __bf16* __restrict__ w_spk, const float* __restrict__ b_spk,
                 const __bf16* __restrict__ w_emo, const float* __restrict__ b_emo,
                 float* __restrict__ out_spk, float* __restrict__ out_emo, int relu) {
  constexpr int KREAL  = CIN * 3;
  constexpr int KPAD   = (KREAL + 31) & ~31;
  constexpr int KSTEPS = KPAD / 32;
  __shared__ __align__(16) __bf16 s_in[CIN * 18];
  const int t0 = blockIdx.x * 16;
  const int b  = blockIdx.y;
  const int e  = blockIdx.z;
  const __bf16* w = e ? w_emo : w_spk;
  const float* bs = e ? b_emo : b_spk;
  const float* in = (e ? in_emo : in_spk) + (size_t)b * CIN * 1000;
  float* out = e ? out_emo : out_spk;
  const int tid = threadIdx.x;
  for (int i = tid; i < CIN * 18; i += 256) {
    const int cin = i / 18, tt = i - cin * 18;
    const int t = t0 - 1 + tt;
    s_in[i] = (t >= 0 && t < 1000) ? (__bf16)in[cin * 1000 + t] : (__bf16)0.f;
  }
  __syncthreads();

  const int lane = tid & 31, half = lane >> 4, lm = lane & 15;
  const int c0 = (tid >> 5) * 16;
  const __bf16* aRow = w + (size_t)(c0 + lm) * KPAD;
  v8f acc = {};
#pragma unroll
  for (int ks = 0; ks < KSTEPS; ++ks) {
    const int kb = ks * 32;
    const v16bf a = frag_a_bf(aRow, kb, half);
    v16bf bf;
#pragma unroll
    for (int j = 0; j < 16; ++j) {
      const int k = kb + half * 16 + j;            // compile-time constant
      if (k < KREAL) {
        const int cin = k / 3, dk = k - cin * 3;   // folds at compile time
        bf[j] = s_in[cin * 18 + lm + dk];
      } else {
        bf[j] = (__bf16)0.f;
      }
    }
    acc = WMMA_BF16(a, bf, acc);
  }
  const int t = t0 + lm;
  if (t >= 1000) return;
  const int mh = half * 8;
#pragma unroll
  for (int r = 0; r < 8; ++r) {
    const int c = c0 + mh + r;
    float v = acc[r] + bs[c];
    out[((size_t)b * 128 + c) * 1000 + t] = relu ? fmaxf(v, 0.f) : v;
  }
}

// ---------------------------------------------------------------------------
// Pool: e==0 mean over T (speaker), e==1 max over T (emotion).
// ---------------------------------------------------------------------------
__global__ void pool_kernel(const float* __restrict__ x2s, const float* __restrict__ x2e,
                            float* __restrict__ pooled) {
  const int b = blockIdx.x, e = blockIdx.y, c = threadIdx.x;  // 128 threads
  const float4* x4 = (const float4*)((e ? x2e : x2s) + ((size_t)b * 128 + c) * 1000);
  float r;
  if (e == 0) {
    float s = 0.f;
    for (int t = 0; t < 250; ++t) { float4 v = x4[t]; s += (v.x + v.y) + (v.z + v.w); }
    r = s * (1.f / 1000.f);
  } else {
    float mx = -INFINITY;
    for (int t = 0; t < 250; ++t) {
      float4 v = x4[t];
      mx = fmaxf(mx, fmaxf(fmaxf(v.x, v.y), fmaxf(v.z, v.w)));
    }
    r = mx;
  }
  pooled[(e * 32 + b) * 128 + c] = r;
}

// ---------------------------------------------------------------------------
// Style FC: se|ee = pooled @ fc_w.T + b
// ---------------------------------------------------------------------------
__global__ void style_kernel(const float* __restrict__ pooled,
                             const float* __restrict__ sfw, const float* __restrict__ sfb,
                             const float* __restrict__ efw, const float* __restrict__ efb,
                             float* __restrict__ style) {
  const int b = blockIdx.x, c = threadIdx.x;  // 256 threads
  const float* pv = pooled + ((c < 128 ? 0 : 32) + b) * 128;
  const float* W  = (c < 128) ? sfw : efw;
  const float* Bv = (c < 128) ? sfb : efb;
  const int cl = c & 127;
  float s = Bv[cl];
  const float4* p4 = (const float4*)pv;
  const float4* w4 = (const float4*)(W + cl * 128);
  for (int k = 0; k < 32; ++k) {
    float4 a = p4[k], w = w4[k];
    s += a.x * w.x + a.y * w.y + a.z * w.z + a.w * w.w;
  }
  style[b * 256 + c] = s;
}

// ---------------------------------------------------------------------------
// Classifier heads -> d_out (sl 32x128, el 32x8).
// ---------------------------------------------------------------------------
__global__ void cls_kernel(const float* __restrict__ style,
                           const float* __restrict__ sw, const float* __restrict__ sb,
                           const float* __restrict__ ew, const float* __restrict__ eb,
                           float* __restrict__ out_sl, float* __restrict__ out_el) {
  const int b = blockIdx.x, j = threadIdx.x;  // 160 threads
  if (j < 128) {
    float s = sb[j];
    for (int k = 0; k < 128; ++k) s += style[b * 256 + k] * sw[j * 128 + k];
    out_sl[b * 128 + j] = s;
  } else if (j < 136) {
    const int jj = j - 128;
    float s = eb[jj];
    for (int k = 0; k < 128; ++k) s += style[b * 256 + 128 + k] * ew[jj * 128 + k];
    out_el[b * 8 + jj] = s;
  }
}

// ---------------------------------------------------------------------------
// Persistent-workgroup LSTM scan (B=32,H=256), bf16 operands. 32 waves =
// (2 batch tiles) x (16 hidden-col tiles); each wave owns all 4 gate tiles of
// its 16x16 h/c tile (c in registers, h via bf16 LDS). A fragments hoisted
// across the 4 gates. Dual parameter sets: blockIdx.x selects direction so
// forward/backward text LSTMs run concurrently on two WGPs.
// ---------------------------------------------------------------------------
template <int KXPAD>
__global__ __launch_bounds__(1024)
void lstm_kernel(const __bf16* __restrict__ x,
                 const __bf16* Wih0, const __bf16* Whh0,
                 const float* bih0, const float* bhh0, __bf16* hout0, int rev0,
                 const __bf16* Wih1, const __bf16* Whh1,
                 const float* bih1, const float* bhh1, __bf16* hout1, int rev1,
                 int T,
                 const float* __restrict__ fc_w, const float* __restrict__ fc_b,
                 float* __restrict__ wout) {
  constexpr int KXSTEPS = KXPAD / 32;
  __shared__ __align__(32) __bf16 h_bf[32 * 256];
  __shared__ __align__(16) float  h_f32[32 * 256];   // only written when fc_w
  const int d = blockIdx.x;
  const __bf16* Wih = d ? Wih1 : Wih0;
  const __bf16* Whh = d ? Whh1 : Whh0;
  const float*  bih = d ? bih1 : bih0;
  const float*  bhh = d ? bhh1 : bhh0;
  __bf16* hout = d ? hout1 : hout0;
  const int reverse = d ? rev1 : rev0;

  const int tid = threadIdx.x;
  for (int i = tid; i < 32 * 256; i += 1024) h_bf[i] = (__bf16)0.f;
  __syncthreads();

  const int wave = tid >> 5;
  const int mt = wave >> 4;          // batch tile 0..1
  const int jt = wave & 15;          // hidden-column tile 0..15
  const int lane = tid & 31, half = lane >> 4, lm = lane & 15;
  const int mh = half * 8;

  const __bf16* wihRow[4];
  const __bf16* whhRow[4];
  float bias[4];
#pragma unroll
  for (int G = 0; G < 4; ++G) {
    const int n = G * 256 + jt * 16 + lm;
    wihRow[G] = Wih + (size_t)n * KXPAD;
    whhRow[G] = Whh + (size_t)n * 256;
    bias[G] = bih[n] + bhh[n];
  }
  const __bf16* aRowH = h_bf + (mt * 16 + lm) * 256;

  float creg[8];
#pragma unroll
  for (int r = 0; r < 8; ++r) creg[r] = 0.f;

  for (int t = 0; t < T; ++t) {
    const int l = reverse ? (T - 1 - t) : t;
    const __bf16* aRowX = x + ((size_t)l * 32 + mt * 16 + lm) * KXPAD;

    v8f accs[4];
#pragma unroll
    for (int G = 0; G < 4; ++G) {
      v8f a;
#pragma unroll
      for (int r = 0; r < 8; ++r) a[r] = bias[G];
      accs[G] = a;
    }
#pragma unroll
    for (int ks = 0; ks < KXSTEPS; ++ks) {        // x @ Wih.T
      const int kb = ks * 32;
      const v16bf a = frag_a_bf(aRowX, kb, half);
#pragma unroll
      for (int G = 0; G < 4; ++G)
        accs[G] = WMMA_BF16(a, frag_b_bf(wihRow[G] + kb + half * 16), accs[G]);
    }
#pragma unroll
    for (int ks = 0; ks < 8; ++ks) {              // h @ Whh.T
      const int kb = ks * 32;
      const v16bf a = frag_a_bf(aRowH, kb, half);
#pragma unroll
      for (int G = 0; G < 4; ++G)
        accs[G] = WMMA_BF16(a, frag_b_bf(whhRow[G] + kb + half * 16), accs[G]);
    }
    __syncthreads();                 // all waves done reading old h

    const int j = jt * 16 + lm;
#pragma unroll
    for (int r = 0; r < 8; ++r) {
      const int bb = mt * 16 + mh + r;
      const float iv = sigf(accs[0][r]);
      const float fv = sigf(accs[1][r]);
      const float gv = tanhf(accs[2][r]);
      const float ov = sigf(accs[3][r]);
      const float cv = fv * creg[r] + iv * gv;
      creg[r] = cv;
      const float hv = ov * tanhf(cv);
      h_bf[bb * 256 + j] = (__bf16)hv;
      if (fc_w) h_f32[bb * 256 + j] = hv;
      if (hout) hout[((size_t)l * 32 + bb) * 256 + j] = (__bf16)hv;
    }
    __syncthreads();                 // new h visible

    if (fc_w && tid < 32) {          // fused vocoder FC: w[b][t]
      float s = fc_b[0];
      const float4* h4 = (const float4*)(h_f32 + tid * 256);
      const float4* w4 = (const float4*)fc_w;
      for (int k = 0; k < 64; ++k) {
        float4 a = h4[k], w = w4[k];
        s += a.x * w.x + a.y * w.y + a.z * w.z + a.w * w.w;
      }
      wout[tid * T + l] = s;
    }
  }
}

// ---------------------------------------------------------------------------
// enc = concat(hf,hb) @ txt_lin_w.T + b : rows = b*512+l (16384), K=512, N=256.
// ---------------------------------------------------------------------------
__global__ __launch_bounds__(256)
void enc_gemm_kernel(const __bf16* __restrict__ hf, const __bf16* __restrict__ hb,
                     const __bf16* __restrict__ lw, const float* __restrict__ lb,
                     float* __restrict__ enc) {
  const int row0 = blockIdx.x * 16;
  const int nt = blockIdx.y * 8 + (threadIdx.x >> 5);
  const int n0 = nt * 16;
  const int lane = threadIdx.x & 31, half = lane >> 4, lm = lane & 15;
  const int row = row0 + lm;           // b*512 + l
  const int bb = row >> 9, l = row & 511;
  const __bf16* aF = hf + ((size_t)l * 32 + bb) * 256;
  const __bf16* aB = hb + ((size_t)l * 32 + bb) * 256 - 256;  // for kb>=256
  const __bf16* bRow = lw + (size_t)(n0 + lm) * 512;
  v8f acc = {};
#pragma unroll
  for (int ks = 0; ks < 16; ++ks) {
    const int kb = ks * 32;
    const __bf16* aRow = (ks < 8) ? aF : aB;
    const v16bf a = frag_a_bf(aRow, kb, half);
    const v16bf b = frag_b_bf(bRow + kb + half * 16);
    acc = WMMA_BF16(a, b, acc);
  }
  const int n = n0 + lm;
  const int mhh = half * 8;
#pragma unroll
  for (int r = 0; r < 8; ++r) {
    const int rr = row0 + mhh + r;
    enc[(size_t)rr * 256 + n] = acc[r] + lb[n];
  }
}

// ---------------------------------------------------------------------------
// Tacotron decoder: one workgroup per batch element, 1000 sequential steps.
// ofs written as bf16 with padded stride 96 (vocoder WMMA A-operand).
// ---------------------------------------------------------------------------
__global__ __launch_bounds__(512)
void decoder_kernel(const float* __restrict__ ref_mel, const float* __restrict__ enc,
                    const float* __restrict__ style,
                    const float* __restrict__ Wih, const float* __restrict__ Whh,
                    const float* __restrict__ bih, const float* __restrict__ bhh,
                    const float* __restrict__ attn_q, const float* __restrict__ attn_e,
                    const float* __restrict__ out_w, const float* __restrict__ out_b,
                    float* __restrict__ out_m, __bf16* __restrict__ ofs) {
  const int b = blockIdx.x;
  const int tid = threadIdx.x;
  __shared__ __align__(16) float h[256], c[256], q[256], xc[336], sa[512], red[512], g[1024];
  if (tid < 256) { h[tid] = 0.f; c[tid] = 0.f; }
  __syncthreads();
  const float* encb = enc + (size_t)b * 512 * 256;
  const float* melb = ref_mel + (size_t)b * 80 * 1000;
  const float* stb  = style + b * 256;

  for (int t = 0; t < 1000; ++t) {
    if (tid < 256) {                        // q = h @ attn_q.T
      float s = 0.f;
      const float4* h4 = (const float4*)h;
      const float4* w4 = (const float4*)(attn_q + tid * 256);
      for (int k = 0; k < 64; ++k) {
        float4 a = h4[k], w = w4[k];
        s += a.x * w.x + a.y * w.y + a.z * w.z + a.w * w.w;
      }
      q[tid] = s;
    }
    if (tid < 80) xc[tid] = (t == 0) ? 0.f : melb[tid * 1000 + (t - 1)];
    __syncthreads();

    {                                       // s[l] = sum tanh(enc+q)*e
      const float4* e4 = (const float4*)(encb + (size_t)tid * 256);
      const float4* q4 = (const float4*)q;
      const float4* a4 = (const float4*)attn_e;
      __builtin_prefetch(e4, 0, 3);
      float s = 0.f;
      for (int k = 0; k < 64; ++k) {
        float4 ev = e4[k], qv = q4[k], av = a4[k];
        s += tanhf(ev.x + qv.x) * av.x + tanhf(ev.y + qv.y) * av.y +
             tanhf(ev.z + qv.z) * av.z + tanhf(ev.w + qv.w) * av.w;
      }
      sa[tid] = s;
    }
    __syncthreads();
    red[tid] = sa[tid]; __syncthreads();    // softmax: max
    for (int st = 256; st > 0; st >>= 1) {
      if (tid < st) red[tid] = fmaxf(red[tid], red[tid + st]);
      __syncthreads();
    }
    const float mx = red[0]; __syncthreads();
    const float p = __expf(sa[tid] - mx);
    red[tid] = p; __syncthreads();          // softmax: sum
    for (int st = 256; st > 0; st >>= 1) {
      if (tid < st) red[tid] += red[tid + st];
      __syncthreads();
    }
    const float inv = 1.f / red[0]; __syncthreads();
    sa[tid] = p * inv;
    __syncthreads();

    {                                       // context cc[h] = sum a[l]*enc[l][h]
      const int hh = tid >> 1, part = tid & 1;
      const float* ec = encb + (size_t)(part * 256) * 256 + hh;
      float s = 0.f;
      for (int l = 0; l < 256; ++l) s += sa[part * 256 + l] * ec[(size_t)l * 256];
      red[tid] = s;
    }
    __syncthreads();
    if (tid < 256) xc[80 + tid] = red[tid * 2] + red[tid * 2 + 1];
    __syncthreads();

    for (int gi = tid; gi < 1024; gi += 512) {  // gates
      float s = bih[gi] + bhh[gi];
      const float4* x4 = (const float4*)xc;
      const float4* wi4 = (const float4*)(Wih + (size_t)gi * 336);
      for (int k = 0; k < 84; ++k) {
        float4 a = x4[k], w = wi4[k];
        s += a.x * w.x + a.y * w.y + a.z * w.z + a.w * w.w;
      }
      const float4* h4 = (const float4*)h;
      const float4* wh4 = (const float4*)(Whh + (size_t)gi * 256);
      for (int k = 0; k < 64; ++k) {
        float4 a = h4[k], w = wh4[k];
        s += a.x * w.x + a.y * w.y + a.z * w.z + a.w * w.w;
      }
      g[gi] = s;
    }
    __syncthreads();
    if (tid < 256) {
      const float iv = sigf(g[tid]);
      const float fv = sigf(g[256 + tid]);
      const float gv = tanhf(g[512 + tid]);
      const float ov = sigf(g[768 + tid]);
      const float cv = fv * c[tid] + iv * gv;
      c[tid] = cv;
      h[tid] = ov * tanhf(cv);
    }
    __syncthreads();
    if (tid < 96) {                         // output frame (pad ofs cols 80..95)
      float s = 0.f;
      if (tid < 80) {
        s = out_b[tid];
        const float* wr = out_w + tid * 512;
        const float4* h4 = (const float4*)h;
        const float4* s4 = (const float4*)stb;
        const float4* w0 = (const float4*)wr;
        const float4* w1 = (const float4*)(wr + 256);
        for (int k = 0; k < 64; ++k) {
          float4 a = h4[k], w = w0[k];
          s += a.x * w.x + a.y * w.y + a.z * w.z + a.w * w.w;
        }
        for (int k = 0; k < 64; ++k) {
          float4 a = s4[k], w = w1[k];
          s += a.x * w.x + a.y * w.y + a.z * w.z + a.w * w.w;
        }
        out_m[((size_t)b * 80 + tid) * 1000 + t] = s;
      }
      ofs[((size_t)t * 32 + b) * 96 + tid] = (__bf16)s;   // zeros for cols 80..95
    }
    __syncthreads();
  }
}

// ---------------------------------------------------------------------------
// Host launch
// ---------------------------------------------------------------------------
extern "C" void kernel_launch(void* const* d_in, const int* in_sizes, int n_in,
                              void* d_out, int out_size, void* d_ws, size_t ws_size,
                              hipStream_t stream) {
  (void)in_sizes; (void)n_in; (void)out_size; (void)ws_size;
  const float* ref_mel = (const float*)d_in[0];
  const int*   tokens  = (const int*)d_in[1];
  const float* spk_c1_w = (const float*)d_in[2];
  const float* spk_c1_b = (const float*)d_in[3];
  const float* spk_c2_w = (const float*)d_in[4];
  const float* spk_c2_b = (const float*)d_in[5];
  const float* spk_fc_w = (const float*)d_in[6];
  const float* spk_fc_b = (const float*)d_in[7];
  const float* emo_c1_w = (const float*)d_in[8];
  const float* emo_c1_b = (const float*)d_in[9];
  const float* emo_c2_w = (const float*)d_in[10];
  const float* emo_c2_b = (const float*)d_in[11];
  const float* emo_fc_w = (const float*)d_in[12];
  const float* emo_fc_b = (const float*)d_in[13];
  const float* spk_cls_w = (const float*)d_in[14];
  const float* spk_cls_b = (const float*)d_in[15];
  const float* emo_cls_w = (const float*)d_in[16];
  const float* emo_cls_b = (const float*)d_in[17];
  const float* emb = (const float*)d_in[18];
  const float* txt_f_Wih = (const float*)d_in[19];
  const float* txt_f_Whh = (const float*)d_in[20];
  const float* txt_f_bih = (const float*)d_in[21];
  const float* txt_f_bhh = (const float*)d_in[22];
  const float* txt_b_Wih = (const float*)d_in[23];
  const float* txt_b_Whh = (const float*)d_in[24];
  const float* txt_b_bih = (const float*)d_in[25];
  const float* txt_b_bhh = (const float*)d_in[26];
  const float* txt_lin_w = (const float*)d_in[27];
  const float* txt_lin_b = (const float*)d_in[28];
  const float* dec_Wih = (const float*)d_in[29];
  const float* dec_Whh = (const float*)d_in[30];
  const float* dec_bih = (const float*)d_in[31];
  const float* dec_bhh = (const float*)d_in[32];
  const float* attn_q = (const float*)d_in[33];
  const float* attn_e = (const float*)d_in[34];
  const float* out_w = (const float*)d_in[35];
  const float* out_b = (const float*)d_in[36];
  const float* voc_Wih = (const float*)d_in[37];
  const float* voc_Whh = (const float*)d_in[38];
  const float* voc_bih = (const float*)d_in[39];
  const float* voc_bhh = (const float*)d_in[40];
  const float* voc_fc_w = (const float*)d_in[41];
  const float* voc_fc_b = (const float*)d_in[42];

  // workspace layout (bytes, all offsets 32B-aligned)
  char* W = (char*)d_ws;
  float*  enc  = (float*)(W + 0);              // 32*512*256 f32
  float*  x1s  = (float*)(W + 16777216);       // 32*128*1000 f32 each
  float*  x1e  = (float*)(W + 33161216);
  float*  x2s  = (float*)(W + 49545216);
  float*  x2e  = (float*)(W + 65929216);
  float*  pool = (float*)(W + 82313216);       // 2*32*128
  float*  styl = (float*)(W + 82345984);       // 32*256
  __bf16* et   = (__bf16*)(W + 82378752);      // 512*32*256 bf16
  __bf16* hf   = (__bf16*)(W + 90767360);
  __bf16* hb   = (__bf16*)(W + 99155968);
  __bf16* wc1s = (__bf16*)(W + 107544576);     // 128*256
  __bf16* wc1e = (__bf16*)(W + 107610112);     // 128*256
  __bf16* wc2s = (__bf16*)(W + 107675648);     // 128*384
  __bf16* wc2e = (__bf16*)(W + 107773952);     // 128*384
  __bf16* wfih = (__bf16*)(W + 107872256);     // 1024*256
  __bf16* wfhh = (__bf16*)(W + 108396544);
  __bf16* wbih = (__bf16*)(W + 108920832);
  __bf16* wbhh = (__bf16*)(W + 109445120);
  __bf16* wlin = (__bf16*)(W + 109969408);     // 256*512
  __bf16* wvih = (__bf16*)(W + 110231552);     // 1024*96
  __bf16* wvhh = (__bf16*)(W + 110428160);     // 1024*256
  __bf16* ofs  = (__bf16*)(W + 16777216);      // alias x1s (dead by decoder)

  float* out_m  = (float*)d_out;               // (32,80,1000)
  float* out_wv = out_m + 2560000;             // (32,1000)
  float* out_sl = out_m + 2592000;             // (32,128)
  float* out_el = out_m + 2596096;             // (32,8)

  // 0) weight conversions to bf16 (+ zero padding where K % 32 != 0)
  pad_rows_bf16_kernel<<<128, 256, 0, stream>>>(spk_c1_w, wc1s, 240, 256, 32768);
  pad_rows_bf16_kernel<<<128, 256, 0, stream>>>(emo_c1_w, wc1e, 240, 256, 32768);
  pad_rows_bf16_kernel<<<192, 256, 0, stream>>>(spk_c2_w, wc2s, 384, 384, 49152);
  pad_rows_bf16_kernel<<<192, 256, 0, stream>>>(emo_c2_w, wc2e, 384, 384, 49152);
  pad_rows_bf16_kernel<<<1024, 256, 0, stream>>>(txt_f_Wih, wfih, 256, 256, 262144);
  pad_rows_bf16_kernel<<<1024, 256, 0, stream>>>(txt_f_Whh, wfhh, 256, 256, 262144);
  pad_rows_bf16_kernel<<<1024, 256, 0, stream>>>(txt_b_Wih, wbih, 256, 256, 262144);
  pad_rows_bf16_kernel<<<1024, 256, 0, stream>>>(txt_b_Whh, wbhh, 256, 256, 262144);
  pad_rows_bf16_kernel<<<512, 256, 0, stream>>>(txt_lin_w, wlin, 512, 512, 131072);
  pad_rows_bf16_kernel<<<384, 256, 0, stream>>>(voc_Wih, wvih, 80, 96, 98304);
  pad_rows_bf16_kernel<<<1024, 256, 0, stream>>>(voc_Whh, wvhh, 256, 256, 262144);
  // 1) embedding (bf16)
  embed_kernel<<<4096, 256, 0, stream>>>(tokens, emb, et);
  // 2) conv1 (Cin=80, padded K=256)
  conv_kernel<80><<<dim3(63, 32, 2), 256, 0, stream>>>(
      ref_mel, ref_mel, wc1s, spk_c1_b, wc1e, emo_c1_b, x1s, x1e, 0);
  // 3) conv2 + relu (Cin=128, K=384 exact)
  conv_kernel<128><<<dim3(63, 32, 2), 256, 0, stream>>>(
      x1s, x1e, wc2s, spk_c2_b, wc2e, emo_c2_b, x2s, x2e, 1);
  // 4) pooling (mean / max)
  pool_kernel<<<dim3(32, 2), 128, 0, stream>>>(x2s, x2e, pool);
  // 5) style FC
  style_kernel<<<32, 256, 0, stream>>>(pool, spk_fc_w, spk_fc_b,
                                       emo_fc_w, emo_fc_b, styl);
  // 6) classifier heads
  cls_kernel<<<32, 160, 0, stream>>>(styl, spk_cls_w, spk_cls_b,
                                     emo_cls_w, emo_cls_b, out_sl, out_el);
  // 7) text LSTMs: forward (block 0) and backward (block 1) concurrently
  lstm_kernel<256><<<2, 1024, 0, stream>>>(
      et,
      wfih, wfhh, txt_f_bih, txt_f_bhh, hf, 0,
      wbih, wbhh, txt_b_bih, txt_b_bhh, hb, 1,
      512, nullptr, nullptr, nullptr);
  // 8) enc projection
  enc_gemm_kernel<<<dim3(1024, 2), 256, 0, stream>>>(hf, hb, wlin, txt_lin_b, enc);
  // 9) decoder (one workgroup per batch element)
  decoder_kernel<<<32, 512, 0, stream>>>(ref_mel, enc, styl,
                                         dec_Wih, dec_Whh, dec_bih, dec_bhh,
                                         attn_q, attn_e, out_w, out_b,
                                         out_m, ofs);
  // 10) vocoder LSTM (padded K=96) with fused FC -> w
  lstm_kernel<96><<<1, 1024, 0, stream>>>(
      ofs,
      wvih, wvhh, voc_bih, voc_bhh, nullptr, 0,
      nullptr, nullptr, nullptr, nullptr, nullptr, 0,
      1000, voc_fc_w, voc_fc_b, out_wv);
}
